// ImprovedCrossBorderGNN_9526237462757
// MI455X (gfx1250) — matrix operand order repaired and checked by reading, not compiled, not run on hardware
//
#include <hip/hip_runtime.h>
#include <hip/hip_bf16.h>
#include <stdint.h>

#define N_NODES 50000
#define N_EDGES 1600000
#define HDIM    128
#define NFEAT   8
#define EP      (N_EDGES + N_NODES)   // edges + self loops
#define NH      (N_NODES * HDIM)

typedef __bf16 bf16;
typedef __attribute__((ext_vector_type(16))) __bf16 v16bf;
typedef __attribute__((ext_vector_type(8)))  __bf16 v8bf;
typedef __attribute__((ext_vector_type(8)))  float  v8f;

// ---- monotone float<->uint encoding for atomicMax-based segment max ----
__device__ __forceinline__ unsigned fenc(float f) {
  unsigned u = __float_as_uint(f);
  return (u & 0x80000000u) ? ~u : (u | 0x80000000u);
}
__device__ __forceinline__ float fdec(unsigned e) {
  unsigned u = (e & 0x80000000u) ? (e & 0x7fffffffu) : ~e;
  return __uint_as_float(u);
}

// ---------------- one-time kernels ----------------
__global__ void k_zero_scal(float* scal) {
  if (threadIdx.x < 16) scal[threadIdx.x] = 0.f;
}

__global__ void k_mean(const float* __restrict__ ea, float* scal) {
  float s0 = 0.f, s1 = 0.f;
  for (int i = blockIdx.x * blockDim.x + threadIdx.x; i < N_EDGES;
       i += gridDim.x * blockDim.x) {
    s0 += ea[2 * i];
    s1 += ea[2 * i + 1];
  }
  for (int o = 16; o; o >>= 1) {
    s0 += __shfl_down(s0, o, 32);
    s1 += __shfl_down(s1, o, 32);
  }
  if ((threadIdx.x & 31) == 0) {
    atomicAdd(&scal[0], s0);
    atomicAdd(&scal[1], s1);
  }
}

// h0 = relu((x @ enc_W + enc_b) * (bn_g/sqrt(1+eps)) + bn_b)
__global__ void k_encoder(const float* __restrict__ x, const float* __restrict__ Wenc,
                          const float* __restrict__ benc, const float* __restrict__ bng,
                          const float* __restrict__ bnb, float* __restrict__ h0) {
  __shared__ float xr[NFEAT];
  int n = blockIdx.x, k = threadIdx.x;
  if (k < NFEAT) xr[k] = x[n * NFEAT + k];
  __syncthreads();
  float s = benc[k];
#pragma unroll
  for (int j = 0; j < NFEAT; j++) s += xr[j] * Wenc[j * HDIM + k];
  const float inv = 0.99999500003749969f;  // 1/sqrt(1+1e-5)
  float v = s * (bng[k] * inv) + bnb[k];
  h0[n * HDIM + k] = v > 0.f ? v : 0.f;
}

// ---------------- per-layer kernels ----------------
__global__ void k_f2bf(const float* __restrict__ src, bf16* __restrict__ dst, int n) {
  for (int i = blockIdx.x * blockDim.x + threadIdx.x; i < n;
       i += gridDim.x * blockDim.x)
    dst[i] = (bf16)src[i];
}

// transpose W [128k x 128n] -> Wt bf16 [col][k] so B columns are contiguous
__global__ void k_wt2bf(const float* __restrict__ W, bf16* __restrict__ wt) {
  int i = blockIdx.x * blockDim.x + threadIdx.x;
  if (i < HDIM * HDIM) {
    int col = i >> 7, k = i & 127;
    wt[i] = (bf16)W[k * HDIM + col];
  }
}

// g = h @ W : bf16 inputs, f32 accumulate, V_WMMA_F32_16X16X32_BF16.
// One wave = 16 rows x 128 cols. Whole Wt staged in LDS (32 KB) via
// GLOBAL_LOAD_ASYNC_TO_LDS_B128 (ASYNCcnt-tracked DMA, no VGPR round-trip).
__global__ void __launch_bounds__(256)
k_gemm(const bf16* __restrict__ hbf, const bf16* __restrict__ wt,
       float* __restrict__ g) {
  __shared__ bf16 sW[HDIM * HDIM];
  {
    // low 32 bits of a flat shared-aperture address are the LDS byte offset
    uint32_t ldsBase = (uint32_t)(uintptr_t)(&sW[0]);
    const char* gsrc = (const char*)wt;
    int t = threadIdx.x;
#pragma unroll
    for (int it = 0; it < 8; it++) {  // 256 threads * 16B * 8 = 32 KB
      uint32_t loff  = ldsBase + (uint32_t)(t * 16 + it * 4096);
      uint64_t gaddr = (uint64_t)(uintptr_t)(gsrc + t * 16 + it * 4096);
      asm volatile("global_load_async_to_lds_b128 %0, %1, off"
                   :: "v"(loff), "v"(gaddr) : "memory");
    }
    asm volatile("s_wait_asynccnt 0x0" ::: "memory");
  }
  __syncthreads();

  int wave = threadIdx.x >> 5;
  int lane = threadIdx.x & 31;
  int l16  = lane & 15;
  int hi   = (lane >> 4) & 1;
  int rowbase = blockIdx.x * 128 + wave * 16;
  if (rowbase >= N_NODES) return;
  int arow = rowbase + l16;
  if (arow >= N_NODES) arow = N_NODES - 1;  // clamped load; store is guarded

  v8f zero = {0.f, 0.f, 0.f, 0.f, 0.f, 0.f, 0.f, 0.f};
  v8f acc[8];
#pragma unroll
  for (int nt = 0; nt < 8; nt++) acc[nt] = zero;

  const bf16* hrow = hbf + (size_t)arow * HDIM;
#pragma unroll
  for (int ks = 0; ks < 4; ks++) {
    // A 16x32 bf16: lane(l16,hi) holds K = [k0..k0+15]
    int k0 = ks * 32 + hi * 16;
    v16bf a = *(const v16bf*)(hrow + k0);
#pragma unroll
    for (int nt = 0; nt < 8; nt++) {
      // B 32x16 bf16: lane holds 16 contiguous K for column l16 (half hi)
      int col = nt * 16 + l16;
      v16bf b = *(const v16bf*)(sW + col * HDIM + ks * 32 + hi * 16);
      acc[nt] = __builtin_amdgcn_wmma_f32_16x16x32_bf16(
          false, a, false, b, (short)0, acc[nt], false, false);
    }
  }
  // C/D: VGPR r -> row r (lanes 0-15) / row r+8 (lanes 16-31), col = lane&15
#pragma unroll
  for (int nt = 0; nt < 8; nt++) {
    int colg = nt * 16 + l16;
#pragma unroll
    for (int r = 0; r < 8; r++) {
      int row = rowbase + r + hi * 8;
      if (row < N_NODES) g[(size_t)row * HDIM + colg] = acc[nt][r];
    }
  }
}

// a_src[n] = g[n].a_s ; a_dst[n] = g[n].a_d  (one wave per node, float4/lane)
__global__ void k_dots(const float* __restrict__ g, const float* __restrict__ as_,
                       const float* __restrict__ ad_, float* __restrict__ osrc,
                       float* __restrict__ odst) {
  int w = (blockIdx.x * blockDim.x + threadIdx.x) >> 5;
  int lane = threadIdx.x & 31;
  if (w >= N_NODES) return;
  const float4 gv = *(const float4*)(g + (size_t)w * HDIM + lane * 4);
  const float4 sv = *(const float4*)(as_ + lane * 4);
  const float4 dv = *(const float4*)(ad_ + lane * 4);
  float p = gv.x * sv.x + gv.y * sv.y + gv.z * sv.z + gv.w * sv.w;
  float q = gv.x * dv.x + gv.y * dv.y + gv.z * dv.z + gv.w * dv.w;
  for (int o = 16; o; o >>= 1) {
    p += __shfl_down(p, o, 32);
    q += __shfl_down(q, o, 32);
  }
  if (lane == 0) { osrc[w] = p; odst[w] = q; }
}

// wa = We^T @ a_e (2 scalars) and self-loop edge term from mean(edge_attr)
__global__ void k_edgevec(const float* __restrict__ We, const float* __restrict__ ae,
                          float* scal) {
  __shared__ float r0[HDIM], r1[HDIM];
  int t = threadIdx.x;
  r0[t] = We[t] * ae[t];
  r1[t] = We[HDIM + t] * ae[t];
  __syncthreads();
  for (int s = 64; s; s >>= 1) {
    if (t < s) { r0[t] += r0[t + s]; r1[t] += r1[t + s]; }
    __syncthreads();
  }
  if (t == 0) {
    float wa0 = r0[0], wa1 = r1[0];
    scal[2] = wa0;
    scal[3] = wa1;
    scal[4] = (scal[0] / (float)N_EDGES) * wa0 + (scal[1] / (float)N_EDGES) * wa1;
  }
}

__global__ void k_init(float* __restrict__ accum, unsigned* __restrict__ menc,
                       float* __restrict__ ssum) {
  for (int i = blockIdx.x * blockDim.x + threadIdx.x; i < NH;
       i += gridDim.x * blockDim.x) {
    accum[i] = 0.f;
    if (i < N_NODES) { menc[i] = 0u; ssum[i] = 0.f; }
  }
}

// logits + segment-max via encoded atomicMax
__global__ void k_logits(const int* __restrict__ ei, const float* __restrict__ ea,
                         const float* __restrict__ asrcv, const float* __restrict__ adstv,
                         const float* __restrict__ scal, float* __restrict__ elog,
                         unsigned* __restrict__ menc) {
  float wa0 = scal[2], wa1 = scal[3], lterm = scal[4];
  for (int i = blockIdx.x * blockDim.x + threadIdx.x; i < EP;
       i += gridDim.x * blockDim.x) {
    int s, d;
    float aed;
    if (i < N_EDGES) {
      s = ei[i];
      d = ei[N_EDGES + i];
      aed = ea[2 * i] * wa0 + ea[2 * i + 1] * wa1;
    } else {
      s = d = i - N_EDGES;
      aed = lterm;
    }
    float l = asrcv[s] + adstv[d] + aed;
    l = l > 0.f ? l : 0.2f * l;  // leaky_relu(0.2)
    elog[i] = l;
    atomicMax(&menc[d], fenc(l));
  }
}

// e = exp(logit - max); segment-sum
__global__ void k_expsum(const int* __restrict__ ei, const unsigned* __restrict__ menc,
                         float* __restrict__ elog, float* __restrict__ ssum) {
  for (int i = blockIdx.x * blockDim.x + threadIdx.x; i < EP;
       i += gridDim.x * blockDim.x) {
    int d = (i < N_EDGES) ? ei[N_EDGES + i] : (i - N_EDGES);
    float e = __expf(elog[i] - fdec(menc[d]));
    elog[i] = e;
    atomicAdd(&ssum[d], e);
  }
}

// accum[dst] += alpha * g[src]   (one wave per edge; float4 per lane = 128)
__global__ void __launch_bounds__(256)
k_scatter(const int* __restrict__ ei, const float* __restrict__ elog,
          const float* __restrict__ ssum, const float* __restrict__ g,
          float* __restrict__ accum) {
  int w = blockIdx.x * 8 + (threadIdx.x >> 5);
  if (w >= EP) return;
  int lane = threadIdx.x & 31;
  int s, d;
  if (w < N_EDGES) { s = ei[w]; d = ei[N_EDGES + w]; }
  else             { s = d = w - N_EDGES; }
  float alpha = elog[w] / (ssum[d] + 1e-16f);
  const float4 gv = *(const float4*)(g + (size_t)s * HDIM + lane * 4);
  float* ap = accum + (size_t)d * HDIM + lane * 4;
  atomicAdd(ap + 0, gv.x * alpha);
  atomicAdd(ap + 1, gv.y * alpha);
  atomicAdd(ap + 2, gv.z * alpha);
  atomicAdd(ap + 3, gv.w * alpha);
}

// in-place epilogue: accum = [relu](accum + b [+ add])
__global__ void k_fin(float* __restrict__ accum, const float* __restrict__ b,
                      const float* __restrict__ add, int do_relu) {
  for (int i = blockIdx.x * blockDim.x + threadIdx.x; i < NH;
       i += gridDim.x * blockDim.x) {
    float v = accum[i] + b[i & 127];
    if (add) v += add[i];
    if (do_relu) v = v > 0.f ? v : 0.f;
    accum[i] = v;
  }
}

extern "C" void kernel_launch(void* const* d_in, const int* in_sizes, int n_in,
                              void* d_out, int out_size, void* d_ws, size_t ws_size,
                              hipStream_t stream) {
  (void)in_sizes; (void)n_in; (void)out_size; (void)ws_size;
  const float* x    = (const float*)d_in[0];
  const int*   ei   = (const int*)d_in[1];
  const float* ea   = (const float*)d_in[2];
  const float* Wenc = (const float*)d_in[3];
  const float* benc = (const float*)d_in[4];
  const float* bng  = (const float*)d_in[5];
  const float* bnb  = (const float*)d_in[6];
  const float* Wl[3]  = {(const float*)d_in[7],  (const float*)d_in[13], (const float*)d_in[19]};
  const float* asl[3] = {(const float*)d_in[8],  (const float*)d_in[14], (const float*)d_in[20]};
  const float* adl[3] = {(const float*)d_in[9],  (const float*)d_in[15], (const float*)d_in[21]};
  const float* Wel[3] = {(const float*)d_in[10], (const float*)d_in[16], (const float*)d_in[22]};
  const float* ael[3] = {(const float*)d_in[11], (const float*)d_in[17], (const float*)d_in[23]};
  const float* bl[3]  = {(const float*)d_in[12], (const float*)d_in[18], (const float*)d_in[24]};

  char* ws = (char*)d_ws;
  size_t off = 0;
  auto carve = [&](size_t bytes) {
    char* p = ws + off;
    off = (off + bytes + 255) & ~(size_t)255;
    return p;
  };
  float*    h0    = (float*)carve((size_t)NH * 4);
  float*    hA    = (float*)carve((size_t)NH * 4);
  float*    g     = (float*)carve((size_t)NH * 4);
  bf16*     hbf   = (bf16*)carve((size_t)NH * 2);
  bf16*     wtbf  = (bf16*)carve((size_t)HDIM * HDIM * 2);
  float*    asrcv = (float*)carve((size_t)N_NODES * 4);
  float*    adstv = (float*)carve((size_t)N_NODES * 4);
  unsigned* menc  = (unsigned*)carve((size_t)N_NODES * 4);
  float*    ssum  = (float*)carve((size_t)N_NODES * 4);
  float*    elog  = (float*)carve((size_t)EP * 4);
  float*    scal  = (float*)carve(64);

  // once per call
  k_zero_scal<<<1, 32, 0, stream>>>(scal);
  k_mean<<<2048, 256, 0, stream>>>(ea, scal);
  k_encoder<<<N_NODES, HDIM, 0, stream>>>(x, Wenc, benc, bng, bnb, h0);

  // layer in/accum recycling: g captures all state after GEMM+dots,
  // so layer 2 accumulates into its own input buffer; layer 3 into d_out.
  const float* hin[3] = {h0, hA, hA};
  float*       acc[3] = {hA, hA, (float*)d_out};

  for (int l = 0; l < 3; l++) {
    k_f2bf<<<2048, 256, 0, stream>>>(hin[l], hbf, NH);
    k_wt2bf<<<(HDIM * HDIM + 255) / 256, 256, 0, stream>>>(Wl[l], wtbf);
    k_gemm<<<(N_NODES + 127) / 128, 256, 0, stream>>>(hbf, wtbf, g);
    k_dots<<<(N_NODES * 32 + 255) / 256, 256, 0, stream>>>(g, asl[l], adl[l], asrcv, adstv);
    k_edgevec<<<1, HDIM, 0, stream>>>(Wel[l], ael[l], scal);
    k_init<<<2048, 256, 0, stream>>>(acc[l], menc, ssum);
    k_logits<<<4096, 256, 0, stream>>>(ei, ea, asrcv, adstv, scal, elog, menc);
    k_expsum<<<4096, 256, 0, stream>>>(ei, menc, elog, ssum);
    k_scatter<<<(EP + 7) / 8, 256, 0, stream>>>(ei, elog, ssum, g, acc[l]);
    k_fin<<<2048, 256, 0, stream>>>(acc[l], bl[l], (l == 2) ? h0 : nullptr, (l < 2) ? 1 : 0);
  }
}
